// MultiHeadAttention_9929964389163
// MI455X (gfx1250) — matrix operand order repaired and checked by reading, not compiled, not run on hardware
//
#include <hip/hip_runtime.h>
#include <hip/hip_bf16.h>

// ---------------------------------------------------------------------------
// MultiHeadAttention for MI455X (gfx1250, wave32, WMMA).
//   k1: QKV GEMM  (f32 x,W -> f16 Q/K/V in [B,H,S,HD])
//   k2: flash attention (f16 WMMA, online softmax, O f16 in [B,S,D])
//   k3: out projection (f16 A, f32 out)
// All operand fragments are per-lane contiguous -> ds_load_b128 only.
// K tile in attention staged via global_load_async_to_lds_b128 (ASYNCcnt).
// ---------------------------------------------------------------------------

typedef __attribute__((ext_vector_type(16))) _Float16 v16h;
typedef __attribute__((ext_vector_type(8)))  _Float16 v8h;
typedef __attribute__((ext_vector_type(4)))  _Float16 v4h;
typedef __attribute__((ext_vector_type(8)))  float    v8f;

#define WMMA_F16(A, B, C) \
  __builtin_amdgcn_wmma_f32_16x16x32_f16(false, (A), false, (B), (short)0, (C), false, false)

static constexpr int Bc = 2, Sc = 4096, Dc = 768, Hc = 12, HDc = 64;
static constexpr float SCALE = 0.125f;  // 64^-0.5

// ---- fragment loaders (layouts per CDNA5 ISA §7.12.2) ----------
// A 16x32: lane row m=lane%16, halfs [m*ld + hl*8 .. +8) and [.. +16 .. +8)
__device__ __forceinline__ v16h ld_afrag(const _Float16* src, int ld) {
  const int lane = threadIdx.x & 31;
  const _Float16* p = src + (lane & 15) * ld + ((lane >> 4) << 3);
  const v8h lo = *(const v8h*)p;
  const v8h hi = *(const v8h*)(p + 16);
  return __builtin_shufflevector(lo, hi, 0, 1, 2, 3, 4, 5, 6, 7, 8, 9, 10, 11,
                                 12, 13, 14, 15);
}
// B 32x16 from TRANSPOSED source srcT[n][k]: lane col n=lane%16,
// k = hl*8 + 0..7 and 16 + hl*8 + 0..7.
__device__ __forceinline__ v16h ld_bfragT(const _Float16* srcT, int ld) {
  const int lane = threadIdx.x & 31;
  const _Float16* p = srcT + (lane & 15) * ld + ((lane >> 4) << 3);
  const v8h lo = *(const v8h*)p;
  const v8h hi = *(const v8h*)(p + 16);
  return __builtin_shufflevector(lo, hi, 0, 1, 2, 3, 4, 5, 6, 7, 8, 9, 10, 11,
                                 12, 13, 14, 15);
}

// ---- CDNA5 async global->LDS copy (16B per lane), tracked by ASYNCcnt -----
__device__ __forceinline__ void async_ld_b128(const void* lds_ptr, const void* gptr) {
  // Flat LDS pointer: addr[31:0] is the LDS offset (ISA §10.2 aperture rules).
  const unsigned loff = (unsigned)(unsigned long long)lds_ptr;
  const unsigned long long ga = (unsigned long long)gptr;
  asm volatile("global_load_async_to_lds_b128 %0, %1, off" ::"v"(loff), "v"(ga)
               : "memory");
}
__device__ __forceinline__ void wait_async0() {
  asm volatile("s_wait_asynccnt 0" ::: "memory");
}

// ---------------------------------------------------------------------------
// Kernel 1: QKV = X @ W + b -> f16 Q/K/V in [B,H,S,HD].
// 256 thr (8 waves, 2x4), tile 64(M) x 128(N), K-step 64.
// ---------------------------------------------------------------------------
__global__ __launch_bounds__(256) void qkv_kernel(
    const float* __restrict__ X, const float* __restrict__ W,
    const float* __restrict__ bias, _Float16* __restrict__ Qh,
    _Float16* __restrict__ Kh, _Float16* __restrict__ Vh) {
  constexpr int LDT = 72, LDW = 3 * Dc;
  __shared__ alignas(16) _Float16 Xs[64 * LDT];   // [m][k]
  __shared__ alignas(16) _Float16 Wt[128 * LDT];  // transposed [n][k]

  const int tid = threadIdx.x;
  const int wave = tid >> 5, lane = tid & 31;
  const int hl = lane >> 4, ln = lane & 15;
  const int wm = wave >> 2, wn = wave & 3;
  const int m0 = blockIdx.y * 64, n0 = blockIdx.x * 128;

  // staging coords
  const int xrow = tid >> 2, xseg = tid & 3;       // X: 16 halfs each
  const int wkc = tid >> 4, wnc = tid & 15;        // W: 4k x 8n block

  v8f acc[2][2] = {};

  for (int k0 = 0; k0 < Dc; k0 += 64) {
    __syncthreads();
    {  // X tile 64x64 f32 -> f16 row-major
      const float* src = X + (size_t)(m0 + xrow) * Dc + k0 + xseg * 16;
      if (k0 + 64 < Dc) __builtin_prefetch(src + 64, 0, 0);
      const float4 f0 = *(const float4*)(src + 0);
      const float4 f1 = *(const float4*)(src + 4);
      const float4 f2 = *(const float4*)(src + 8);
      const float4 f3 = *(const float4*)(src + 12);
      const v8h h0 = {(_Float16)f0.x, (_Float16)f0.y, (_Float16)f0.z, (_Float16)f0.w,
                      (_Float16)f1.x, (_Float16)f1.y, (_Float16)f1.z, (_Float16)f1.w};
      const v8h h1 = {(_Float16)f2.x, (_Float16)f2.y, (_Float16)f2.z, (_Float16)f2.w,
                      (_Float16)f3.x, (_Float16)f3.y, (_Float16)f3.z, (_Float16)f3.w};
      *(v8h*)&Xs[xrow * LDT + xseg * 16 + 0] = h0;
      *(v8h*)&Xs[xrow * LDT + xseg * 16 + 8] = h1;
    }
    {  // W tile 64x128 f32 -> f16 TRANSPOSED in 4k x 8n register blocks
      const float* src = W + (size_t)(k0 + wkc * 4) * LDW + n0 + wnc * 8;
      if (k0 + 64 < Dc) __builtin_prefetch(src + (size_t)64 * LDW, 0, 0);
      float rr[4][8];
#pragma unroll
      for (int i = 0; i < 4; ++i) {
        const float4 lo4 = *(const float4*)(src + (size_t)i * LDW);
        const float4 hi4 = *(const float4*)(src + (size_t)i * LDW + 4);
        rr[i][0] = lo4.x; rr[i][1] = lo4.y; rr[i][2] = lo4.z; rr[i][3] = lo4.w;
        rr[i][4] = hi4.x; rr[i][5] = hi4.y; rr[i][6] = hi4.z; rr[i][7] = hi4.w;
      }
#pragma unroll
      for (int j = 0; j < 8; ++j) {
        const v4h c = {(_Float16)rr[0][j], (_Float16)rr[1][j], (_Float16)rr[2][j],
                       (_Float16)rr[3][j]};
        *(v4h*)&Wt[(wnc * 8 + j) * LDT + wkc * 4] = c;
      }
    }
    __syncthreads();

#pragma unroll
    for (int kc = 0; kc < 64; kc += 32) {
      const v16h a0 = ld_afrag(&Xs[(wm * 32 + 0) * LDT + kc], LDT);
      const v16h a1 = ld_afrag(&Xs[(wm * 32 + 16) * LDT + kc], LDT);
      const v16h b0 = ld_bfragT(&Wt[(wn * 32 + 0) * LDT + kc], LDT);
      const v16h b1 = ld_bfragT(&Wt[(wn * 32 + 16) * LDT + kc], LDT);
      acc[0][0] = WMMA_F16(a0, b0, acc[0][0]);
      acc[0][1] = WMMA_F16(a0, b1, acc[0][1]);
      acc[1][0] = WMMA_F16(a1, b0, acc[1][0]);
      acc[1][1] = WMMA_F16(a1, b1, acc[1][1]);
    }
  }

  // Epilogue: bias + scatter to Q/K/V [B,H,S,HD] as f16.
#pragma unroll
  for (int jj = 0; jj < 2; ++jj) {
    const int col = n0 + wn * 32 + jj * 16 + ln;  // 0..2303
    const int t = col / Dc;
    const int rem = col - t * Dc;
    const int h = rem >> 6, hd = rem & 63;
    _Float16* dstbuf = (t == 0) ? Qh : (t == 1) ? Kh : Vh;
    const float bv = bias[col];
#pragma unroll
    for (int ii = 0; ii < 2; ++ii) {
#pragma unroll
      for (int r = 0; r < 8; ++r) {
        const int row = m0 + wm * 32 + ii * 16 + hl * 8 + r;  // 0..8191
        const int b_idx = row >> 12, s_idx = row & 4095;
        dstbuf[((size_t)(b_idx * Hc + h) * Sc + s_idx) * HDc + hd] =
            (_Float16)(acc[ii][jj][r] + bv);
      }
    }
  }
}

// ---------------------------------------------------------------------------
// Kernel 2: flash attention. 256 thr (8 waves), 128 queries/block.
// K tile async-copied to LDS; V tile staged transposed ([hd][key]).
// ---------------------------------------------------------------------------
__global__ __launch_bounds__(256) void attn_kernel(
    const _Float16* __restrict__ Qh, const _Float16* __restrict__ Kh,
    const _Float16* __restrict__ Vh, _Float16* __restrict__ Oh) {
  constexpr int LDT = 72;
  __shared__ alignas(16) _Float16 Ks[64 * LDT];       // [key][hd]
  __shared__ alignas(16) _Float16 Vt[64 * LDT];       // transposed [hd][key]
  __shared__ alignas(16) _Float16 Ps[8 * 16 * LDT];   // per-wave P [m][key]

  const int tid = threadIdx.x;
  const int wave = tid >> 5, lane = tid & 31;
  const int hl = lane >> 4, ln = lane & 15;
  const int bh = blockIdx.y;  // b*H + h
  const int b_idx = bh / Hc, h = bh - b_idx * Hc;
  const size_t head_off = (size_t)bh * Sc * HDc;
  const _Float16* qbase = Qh + head_off;
  const _Float16* kbase = Kh + head_off;
  const _Float16* vbase = Vh + head_off;

  const int q0 = blockIdx.x * 128 + wave * 16;

  v16h aq[2];
  aq[0] = ld_afrag(qbase + (size_t)q0 * HDc + 0, HDc);
  aq[1] = ld_afrag(qbase + (size_t)q0 * HDc + 32, HDc);

  v8f oacc[4] = {};
  float m_row[8], l_row[8];
#pragma unroll
  for (int r = 0; r < 8; ++r) { m_row[r] = -3.0e38f; l_row[r] = 0.0f; }

  _Float16* Pw = &Ps[wave * 16 * LDT];

  // staging coords
  const int krow = tid >> 2, kseg = tid & 3;  // K: 16 halfs each
  const int vk0 = (tid >> 4) * 4, vh0 = (tid & 15) * 4;  // V: 4key x 4hd block

  for (int kb = 0; kb < Sc; kb += 64) {
    __syncthreads();  // prior P@V reads done before restaging K/V
    {  // K tile: async global->LDS, 2 x b128 per thread
      const _Float16* gk = kbase + (size_t)(kb + krow) * HDc + kseg * 16;
      async_ld_b128(&Ks[krow * LDT + kseg * 16 + 0], gk + 0);
      async_ld_b128(&Ks[krow * LDT + kseg * 16 + 8], gk + 8);
      if (kb + 64 < Sc) __builtin_prefetch(gk + 64 * HDc, 0, 0);
    }
    {  // V tile: manual transposed staging (4x4 register block)
      _Float16 vv[4][4];
#pragma unroll
      for (int i = 0; i < 4; ++i) {
        const v4h r = *(const v4h*)(vbase + (size_t)(kb + vk0 + i) * HDc + vh0);
        vv[i][0] = r[0]; vv[i][1] = r[1]; vv[i][2] = r[2]; vv[i][3] = r[3];
      }
#pragma unroll
      for (int j = 0; j < 4; ++j) {
        const v4h c = {vv[0][j], vv[1][j], vv[2][j], vv[3][j]};
        *(v4h*)&Vt[(vh0 + j) * LDT + vk0] = c;
      }
    }
    wait_async0();
    __syncthreads();

    // S = Q @ K^T : 16x64 scores (4 frags), 2 chained WMMA over HD each.
    v8f sf[4];
#pragma unroll
    for (int c = 0; c < 4; ++c) {
      v8f s = {};
      const v16h bk0 = ld_bfragT(&Ks[(c * 16) * LDT + 0], LDT);
      const v16h bk1 = ld_bfragT(&Ks[(c * 16) * LDT + 32], LDT);
      s = WMMA_F16(aq[0], bk0, s);
      s = WMMA_F16(aq[1], bk1, s);
      sf[c] = s;
    }

    // Online softmax; row m = r + hl*8 lives on this half's 16 lanes.
    float mnew[8];
#pragma unroll
    for (int r = 0; r < 8; ++r) {
      float v = fmaxf(fmaxf(sf[0][r], sf[1][r]), fmaxf(sf[2][r], sf[3][r])) * SCALE;
#pragma unroll
      for (int mk = 1; mk <= 8; mk <<= 1) v = fmaxf(v, __shfl_xor(v, mk, 32));
      mnew[r] = fmaxf(m_row[r], v);
    }
    float rs[8];
#pragma unroll
    for (int r = 0; r < 8; ++r) rs[r] = 0.0f;
#pragma unroll
    for (int c = 0; c < 4; ++c) {
#pragma unroll
      for (int r = 0; r < 8; ++r) {
        const float p = __expf(sf[c][r] * SCALE - mnew[r]);
        rs[r] += p;
        Pw[(hl * 8 + r) * LDT + c * 16 + ln] = (_Float16)p;
      }
    }
#pragma unroll
    for (int r = 0; r < 8; ++r) {
      float t = rs[r];
#pragma unroll
      for (int mk = 1; mk <= 8; mk <<= 1) t += __shfl_xor(t, mk, 32);
      const float alpha = __expf(m_row[r] - mnew[r]);
      l_row[r] = l_row[r] * alpha + t;
      m_row[r] = mnew[r];
#pragma unroll
      for (int c2 = 0; c2 < 4; ++c2) oacc[c2][r] *= alpha;
    }
    __syncthreads();

    // O += P @ V (Vt transposed -> contiguous B frags).
    const v16h ap0 = ld_afrag(Pw + 0, LDT);
    const v16h ap1 = ld_afrag(Pw + 32, LDT);
#pragma unroll
    for (int c2 = 0; c2 < 4; ++c2) {
      const v16h bv0 = ld_bfragT(&Vt[(c2 * 16) * LDT + 0], LDT);
      const v16h bv1 = ld_bfragT(&Vt[(c2 * 16) * LDT + 32], LDT);
      oacc[c2] = WMMA_F16(ap0, bv0, oacc[c2]);
      oacc[c2] = WMMA_F16(ap1, bv1, oacc[c2]);
    }
  }

  // Normalize and store O f16 in [B,S,D].
#pragma unroll
  for (int c2 = 0; c2 < 4; ++c2) {
#pragma unroll
    for (int r = 0; r < 8; ++r) {
      const int s_idx = q0 + hl * 8 + r;
      const int col = h * HDc + c2 * 16 + ln;
      Oh[((size_t)b_idx * Sc + s_idx) * Dc + col] =
          (_Float16)(oacc[c2][r] / l_row[r]);
    }
  }
}

// ---------------------------------------------------------------------------
// Kernel 3: out = O @ proj_w + proj_b (f16 A, f32 out). Same skeleton as k1.
// ---------------------------------------------------------------------------
__global__ __launch_bounds__(256) void proj_kernel(
    const _Float16* __restrict__ O, const float* __restrict__ W,
    const float* __restrict__ bias, float* __restrict__ out) {
  constexpr int LDT = 72, LDW = Dc;
  __shared__ alignas(16) _Float16 Xs[64 * LDT];
  __shared__ alignas(16) _Float16 Wt[128 * LDT];

  const int tid = threadIdx.x;
  const int wave = tid >> 5, lane = tid & 31;
  const int hl = lane >> 4, ln = lane & 15;
  const int wm = wave >> 2, wn = wave & 3;
  const int m0 = blockIdx.y * 64, n0 = blockIdx.x * 128;

  const int xrow = tid >> 2, xseg = tid & 3;
  const int wkc = tid >> 4, wnc = tid & 15;

  v8f acc[2][2] = {};

  for (int k0 = 0; k0 < Dc; k0 += 64) {
    __syncthreads();
    {  // A tile (f16): straight vector copy
      const _Float16* src = O + (size_t)(m0 + xrow) * Dc + k0 + xseg * 16;
      if (k0 + 64 < Dc) __builtin_prefetch(src + 64, 0, 0);
      *(v8h*)&Xs[xrow * LDT + xseg * 16 + 0] = *(const v8h*)(src + 0);
      *(v8h*)&Xs[xrow * LDT + xseg * 16 + 8] = *(const v8h*)(src + 8);
    }
    {  // W tile f32 -> f16 transposed
      const float* src = W + (size_t)(k0 + wkc * 4) * LDW + n0 + wnc * 8;
      if (k0 + 64 < Dc) __builtin_prefetch(src + (size_t)64 * LDW, 0, 0);
      float rr[4][8];
#pragma unroll
      for (int i = 0; i < 4; ++i) {
        const float4 lo4 = *(const float4*)(src + (size_t)i * LDW);
        const float4 hi4 = *(const float4*)(src + (size_t)i * LDW + 4);
        rr[i][0] = lo4.x; rr[i][1] = lo4.y; rr[i][2] = lo4.z; rr[i][3] = lo4.w;
        rr[i][4] = hi4.x; rr[i][5] = hi4.y; rr[i][6] = hi4.z; rr[i][7] = hi4.w;
      }
#pragma unroll
      for (int j = 0; j < 8; ++j) {
        const v4h c = {(_Float16)rr[0][j], (_Float16)rr[1][j], (_Float16)rr[2][j],
                       (_Float16)rr[3][j]};
        *(v4h*)&Wt[(wnc * 8 + j) * LDT + wkc * 4] = c;
      }
    }
    __syncthreads();

#pragma unroll
    for (int kc = 0; kc < 64; kc += 32) {
      const v16h a0 = ld_afrag(&Xs[(wm * 32 + 0) * LDT + kc], LDT);
      const v16h a1 = ld_afrag(&Xs[(wm * 32 + 16) * LDT + kc], LDT);
      const v16h b0 = ld_bfragT(&Wt[(wn * 32 + 0) * LDT + kc], LDT);
      const v16h b1 = ld_bfragT(&Wt[(wn * 32 + 16) * LDT + kc], LDT);
      acc[0][0] = WMMA_F16(a0, b0, acc[0][0]);
      acc[0][1] = WMMA_F16(a0, b1, acc[0][1]);
      acc[1][0] = WMMA_F16(a1, b0, acc[1][0]);
      acc[1][1] = WMMA_F16(a1, b1, acc[1][1]);
    }
  }

#pragma unroll
  for (int jj = 0; jj < 2; ++jj) {
    const int col = n0 + wn * 32 + jj * 16 + ln;
    const float bv = bias[col];
#pragma unroll
    for (int ii = 0; ii < 2; ++ii) {
#pragma unroll
      for (int r = 0; r < 8; ++r) {
        const int row = m0 + wm * 32 + ii * 16 + hl * 8 + r;
        out[(size_t)row * Dc + col] = acc[ii][jj][r] + bv;
      }
    }
  }
}

// ---------------------------------------------------------------------------
extern "C" void kernel_launch(void* const* d_in, const int* in_sizes, int n_in,
                              void* d_out, int out_size, void* d_ws, size_t ws_size,
                              hipStream_t stream) {
  const float* x      = (const float*)d_in[0];
  const float* qkv_w  = (const float*)d_in[1];
  const float* qkv_b  = (const float*)d_in[2];
  const float* proj_w = (const float*)d_in[3];
  const float* proj_b = (const float*)d_in[4];
  float* out = (float*)d_out;

  const size_t n_head = (size_t)Bc * Hc * Sc * HDc;  // 6291456
  _Float16* ws = (_Float16*)d_ws;
  _Float16* Qh = ws;
  _Float16* Kh = ws + n_head;
  _Float16* Vh = ws + 2 * n_head;
  _Float16* Oh = ws + 3 * n_head;

  dim3 blk(256);
  qkv_kernel<<<dim3(18, 128), blk, 0, stream>>>(x, qkv_w, qkv_b, Qh, Kh, Vh);
  attn_kernel<<<dim3(Sc / 128, Bc * Hc), blk, 0, stream>>>(Qh, Kh, Vh, Oh);
  proj_kernel<<<dim3(6, 128), blk, 0, stream>>>(Oh, proj_w, proj_b, out);
}
